// AttentionHead_12025908428856
// MI455X (gfx1250) — matrix operand order repaired and checked by reading, not compiled, not run on hardware
//
#include <hip/hip_runtime.h>

typedef __attribute__((ext_vector_type(16))) _Float16 v16h;
typedef __attribute__((ext_vector_type(8)))  _Float16 v8h;
typedef __attribute__((ext_vector_type(8)))  float    v8f;

#define B_ 8
#define T_ 2048
#define C_ 1024
#define H_ 64

union V16 { v16h v; v8h h[2]; };

// ---------------------------------------------------------------------------
// Kernel 0: transpose + f32->f16 convert the three weight matrices.
// Wt layout: [3][H][C]: WMMA B-operand loads (fixed h, varying c) become
// single contiguous 32-byte v16h loads.
// ---------------------------------------------------------------------------
__global__ void wt_kernel(const float* __restrict__ Wq, const float* __restrict__ Wk,
                          const float* __restrict__ Wv, _Float16* __restrict__ Wt) {
    int idx = blockIdx.x * blockDim.x + threadIdx.x;
    if (idx >= 3 * C_ * H_) return;
    int m   = idx / (C_ * H_);
    int rem = idx - m * (C_ * H_);
    int c   = rem / H_;
    int h   = rem - c * H_;
    const float* W = (m == 0) ? Wq : (m == 1) ? Wk : Wv;
    Wt[(size_t)m * H_ * C_ + (size_t)h * C_ + c] = (_Float16)W[(size_t)c * H_ + h];
}

// ----- helpers for the pipelined projection loop -----
__device__ __forceinline__ void loadAraw(const float* pA0, const float* pA1, int c0,
                                         float4& f0, float4& f1, float4& f2, float4& f3) {
    f0 = ((const float4*)(pA0 + c0))[0];
    f1 = ((const float4*)(pA0 + c0))[1];
    f2 = ((const float4*)(pA1 + c0))[0];
    f3 = ((const float4*)(pA1 + c0))[1];
}

__device__ __forceinline__ V16 cvtA(float4 f0, float4 f1, float4 f2, float4 f3) {
    V16 a;
    a.v[0]  = (_Float16)f0.x; a.v[1]  = (_Float16)f0.y;
    a.v[2]  = (_Float16)f0.z; a.v[3]  = (_Float16)f0.w;
    a.v[4]  = (_Float16)f1.x; a.v[5]  = (_Float16)f1.y;
    a.v[6]  = (_Float16)f1.z; a.v[7]  = (_Float16)f1.w;
    a.v[8]  = (_Float16)f2.x; a.v[9]  = (_Float16)f2.y;
    a.v[10] = (_Float16)f2.z; a.v[11] = (_Float16)f2.w;
    a.v[12] = (_Float16)f3.x; a.v[13] = (_Float16)f3.y;
    a.v[14] = (_Float16)f3.z; a.v[15] = (_Float16)f3.w;
    return a;
}

__device__ __forceinline__ void loadB(const _Float16* W, int lo, int hi, int c0, v16h b[4]) {
#pragma unroll
    for (int t = 0; t < 4; ++t)
        b[t] = *(const v16h*)(W + (size_t)(t * 16 + lo) * C_ + c0 + hi * 16);
}

// ---------------------------------------------------------------------------
// Kernel 1: projections.  One wave computes a 16x64 tile of Q, K or V.
//   grid.x = (B*T)/16 row tiles, grid.y = matrix (0=Q,1=K,2=V), block = 32.
// Unroll-by-2 software pipeline: two disjoint register sets alternate, loads
// target the same variables across the backedge (no rotate copies, no spills
// thanks to __launch_bounds__ widening the VGPR budget).
// Q,K stored [B*T,H] f16; V stored transposed [B][H][T] f16.
// Q pre-scaled by C^-0.5 = 1/32.
// ---------------------------------------------------------------------------
__global__ __launch_bounds__(32, 1)
void proj_kernel(const float* __restrict__ X, const _Float16* __restrict__ Wt,
                 _Float16* __restrict__ Qh, _Float16* __restrict__ Kh,
                 _Float16* __restrict__ Vt) {
    const int lane = threadIdx.x;
    const int lo = lane & 15, hi = lane >> 4;
    const int row0 = blockIdx.x * 16;
    const int m = blockIdx.y;
    const _Float16* W = Wt + (size_t)m * H_ * C_;

    v8f acc[4];
#pragma unroll
    for (int t = 0; t < 4; ++t)
#pragma unroll
        for (int r = 0; r < 8; ++r) acc[t][r] = 0.0f;

    const float* xrow = X + (size_t)(row0 + lo) * C_;
    const float* pA0 = xrow + hi * 16;       // A elements 0..7  : K = hi*16 + j
    const float* pA1 = xrow + hi * 16 + 8;   // A elements 8..15 : K = hi*16 + 8 + j

    // ---- prologue: set 0 loads for c0 = 0 ----
    float4 r0a, r0b, r0c, r0d;
    v16h  b0[4];
    loadAraw(pA0, pA1, 0, r0a, r0b, r0c, r0d);
    loadB(W, lo, hi, 0, b0);

    for (int c0 = 0; c0 < C_; c0 += 64) {
        // issue set-1 loads (c0+32) before consuming set 0
        float4 r1a, r1b, r1c, r1d;
        v16h  b1[4];
        loadAraw(pA0, pA1, c0 + 32, r1a, r1b, r1c, r1d);
        loadB(W, lo, hi, c0 + 32, b1);

        V16 a0 = cvtA(r0a, r0b, r0c, r0d);
#pragma unroll
        for (int t = 0; t < 4; ++t)
            acc[t] = __builtin_amdgcn_wmma_f32_16x16x32_f16(
                false, a0.v, false, b0[t], (short)0, acc[t], false, false);

        // refill set 0 for c0+64 while set-1 WMMAs run (uniform guard)
        if (c0 + 64 < C_) {
            loadAraw(pA0, pA1, c0 + 64, r0a, r0b, r0c, r0d);
            loadB(W, lo, hi, c0 + 64, b0);
        }

        V16 a1 = cvtA(r1a, r1b, r1c, r1d);
#pragma unroll
        for (int t = 0; t < 4; ++t)
            acc[t] = __builtin_amdgcn_wmma_f32_16x16x32_f16(
                false, a1.v, false, b1[t], (short)0, acc[t], false, false);
    }

    if (m == 0) {            // Q, scaled by C^-0.5
#pragma unroll
        for (int t = 0; t < 4; ++t)
#pragma unroll
            for (int r = 0; r < 8; ++r)
                Qh[(size_t)(row0 + r + 8 * hi) * H_ + t * 16 + lo] =
                    (_Float16)(acc[t][r] * 0.03125f);
    } else if (m == 1) {     // K
#pragma unroll
        for (int t = 0; t < 4; ++t)
#pragma unroll
            for (int r = 0; r < 8; ++r)
                Kh[(size_t)(row0 + r + 8 * hi) * H_ + t * 16 + lo] =
                    (_Float16)acc[t][r];
    } else {                 // V transposed: Vt[b][h][t]
        int bb   = row0 / T_;
        int tloc = row0 - bb * T_;
#pragma unroll
        for (int t = 0; t < 4; ++t) {
            v8h pk;
#pragma unroll
            for (int r = 0; r < 8; ++r) pk[r] = (_Float16)acc[t][r];
            *(v8h*)(Vt + ((size_t)bb * H_ + t * 16 + lo) * T_ + tloc + 8 * hi) = pk;
        }
    }
}

// ---------------------------------------------------------------------------
// Kernel 2: flash attention.  One wave per (batch, 16-query tile).
//   grid.x = T/16, grid.y = B, block = 32.
// Row-sum of P is computed on the matrix pipe (P x ones): every lane then
// holds its rows' denominators directly (no shuffles for the sum).
// ---------------------------------------------------------------------------
__global__ __launch_bounds__(32, 1)
void flash_kernel(const _Float16* __restrict__ Qh, const _Float16* __restrict__ Kh,
                  const _Float16* __restrict__ Vt, float* __restrict__ out) {
    __shared__ __align__(16) _Float16 sP[16 * 32];  // 16 q-rows x 32 k-cols
    const int lane = threadIdx.x;
    const int lo = lane & 15, hi = lane >> 4;
    const int b  = blockIdx.y;
    const int q0 = blockIdx.x * 16;

    // Q A-operands: two 16x32 f16 tiles (h 0..31 and 32..63), held in registers
    const _Float16* qrow = Qh + ((size_t)b * T_ + q0 + lo) * H_;
    V16 qa[2];
#pragma unroll
    for (int c = 0; c < 2; ++c) {
        qa[c].h[0] = *(const v8h*)(qrow + c * 32 + hi * 16);
        qa[c].h[1] = *(const v8h*)(qrow + c * 32 + hi * 16 + 8);
    }

    v16h ones;
#pragma unroll
    for (int i = 0; i < 16; ++i) ones[i] = (_Float16)1.0f;

    v8f o[4];
    v8f lacc;                 // running softmax denominators (same row<->lane map as o)
    float mr[8];
#pragma unroll
    for (int t = 0; t < 4; ++t)
#pragma unroll
        for (int r = 0; r < 8; ++r) o[t][r] = 0.0f;
#pragma unroll
    for (int r = 0; r < 8; ++r) { mr[r] = -1e30f; lacc[r] = 0.0f; }

    for (int k0 = 0; k0 < q0 + 16; k0 += 32) {
        const _Float16* kr0 = Kh + ((size_t)b * T_ + k0 + lo) * H_;       // keys k0..k0+15
        const _Float16* kr1 = kr0 + 16 * H_;                              // keys k0+16..k0+31

        // prefetch next key/value tiles (global_prefetch_b8)
        if (k0 + 32 < q0 + 16) {
            __builtin_prefetch(kr0 + 32 * H_, 0, 3);
            __builtin_prefetch(kr1 + 32 * H_, 0, 3);
            __builtin_prefetch(Vt + ((size_t)b * H_ + (hi * 2 + 0) * 16 + lo) * T_ + k0 + 32, 0, 3);
            __builtin_prefetch(Vt + ((size_t)b * H_ + (hi * 2 + 1) * 16 + lo) * T_ + k0 + 32, 0, 3);
        }

        // ---- S = Q * K^T for 32 keys (two 16x16 score tiles) ----
        v8f s0, s1;
#pragma unroll
        for (int r = 0; r < 8; ++r) { s0[r] = 0.0f; s1[r] = 0.0f; }
#pragma unroll
        for (int c = 0; c < 2; ++c) {
            v16h bk0 = *(const v16h*)(kr0 + c * 32 + hi * 16);
            v16h bk1 = *(const v16h*)(kr1 + c * 32 + hi * 16);
            s0 = __builtin_amdgcn_wmma_f32_16x16x32_f16(
                false, qa[c].v, false, bk0, (short)0, s0, false, false);
            s1 = __builtin_amdgcn_wmma_f32_16x16x32_f16(
                false, qa[c].v, false, bk1, (short)0, s1, false, false);
        }

        // ---- causal mask + online softmax (D layout: row = r + 8*hi, col = lo) ----
#pragma unroll
        for (int r = 0; r < 8; ++r) {
            int qg = q0 + r + 8 * hi;
            float v0 = ((k0 + lo)      <= qg) ? s0[r] : -1e30f;
            float v1 = ((k0 + 16 + lo) <= qg) ? s1[r] : -1e30f;
            float mx = fmaxf(v0, v1);
            mx = fmaxf(mx, __shfl_xor(mx, 1, 32));
            mx = fmaxf(mx, __shfl_xor(mx, 2, 32));
            mx = fmaxf(mx, __shfl_xor(mx, 4, 32));
            mx = fmaxf(mx, __shfl_xor(mx, 8, 32));
            float mnew = fmaxf(mr[r], mx);
            float sc   = __expf(mr[r] - mnew);
            mr[r] = mnew;
            float p0 = __expf(v0 - mnew);
            float p1 = __expf(v1 - mnew);
            lacc[r] *= sc;
            o[0][r] *= sc; o[1][r] *= sc; o[2][r] *= sc; o[3][r] *= sc;
            sP[(r + 8 * hi) * 32 + lo]      = (_Float16)p0;
            sP[(r + 8 * hi) * 32 + 16 + lo] = (_Float16)p1;
        }
        __syncthreads();

        // ---- P (A layout via LDS transpose): O += P*V, lacc += P*ones ----
        V16 pa;
        pa.h[0] = *(const v8h*)&sP[lo * 32 + hi * 16];
        pa.h[1] = *(const v8h*)&sP[lo * 32 + hi * 16 + 8];
        lacc = __builtin_amdgcn_wmma_f32_16x16x32_f16(
            false, pa.v, false, ones, (short)0, lacc, false, false);
#pragma unroll
        for (int t = 0; t < 4; ++t) {
            v16h bv = *(const v16h*)(Vt + ((size_t)b * H_ + t * 16 + lo) * T_ + k0 + hi * 16);
            o[t] = __builtin_amdgcn_wmma_f32_16x16x32_f16(
                false, pa.v, false, bv, (short)0, o[t], false, false);
        }
        __syncthreads();
    }

    // ---- normalize and store f32 out[b, q, h] ----
#pragma unroll
    for (int t = 0; t < 4; ++t)
#pragma unroll
        for (int r = 0; r < 8; ++r)
            out[((size_t)b * T_ + q0 + r + 8 * hi) * H_ + t * 16 + lo] = o[t][r] / lacc[r];
}

// ---------------------------------------------------------------------------
extern "C" void kernel_launch(void* const* d_in, const int* in_sizes, int n_in,
                              void* d_out, int out_size, void* d_ws, size_t ws_size,
                              hipStream_t stream) {
    const float* X  = (const float*)d_in[0];
    const float* Wq = (const float*)d_in[1];
    const float* Wk = (const float*)d_in[2];
    const float* Wv = (const float*)d_in[3];
    float* out = (float*)d_out;

    char* ws = (char*)d_ws;
    const size_t QB = (size_t)B_ * T_ * H_ * 2;  // 2 MB per f16 [B*T,H] buffer
    _Float16* Qh = (_Float16*)(ws);
    _Float16* Kh = (_Float16*)(ws + QB);
    _Float16* Vt = (_Float16*)(ws + 2 * QB);
    _Float16* Wt = (_Float16*)(ws + 3 * QB);      // 384 KB

    int n = 3 * C_ * H_;
    wt_kernel<<<(n + 255) / 256, 256, 0, stream>>>(Wq, Wk, Wv, Wt);
    proj_kernel<<<dim3((B_ * T_) / 16, 3), 32, 0, stream>>>(X, Wt, Qh, Kh, Vt);
    flash_kernel<<<dim3(T_ / 16, B_), 32, 0, stream>>>(Qh, Kh, Vt, out);
}
